// Fusion_Attention_1168231104899
// MI455X (gfx1250) — matrix operand order repaired and checked by reading, not compiled, not run on hardware
//
#include <hip/hip_runtime.h>
#include <hip/hip_bf16.h>
#include <math.h>

typedef __attribute__((ext_vector_type(16))) _Float16 v16h;
typedef __attribute__((ext_vector_type(8)))  _Float16 v8h;
typedef __attribute__((ext_vector_type(4)))  _Float16 v4h;
typedef __attribute__((ext_vector_type(8)))  float    v8f;
typedef __attribute__((ext_vector_type(4)))  float    v4f;

#define TILE_M 64
#define TILE_N 64
#define TILE_K 32
#define APAD 8   // pad to 40 halves = 80 B rows (16-B aligned, conflict-spreading)

// ---------------------------------------------------------------------------
// Generic GEMM: C[b] = A (MxK weights, shared) * B[b] (KxN activations)
// f32 in/out, f16 WMMA (16x16x32) with f32 accumulate. 128 threads = 4 waves.
// A staged row-major [M][K]; B staged TRANSPOSED [N][K] so the WMMA B-fragment
// (lane n = l&15, k = 8*(l>>4)..+7 and 16+8*(l>>4)..+7) is two ds_load_b128s.
// Staging uses float4 global loads (global_load_b128) for the streaming path.
// ---------------------------------------------------------------------------
__launch_bounds__(128)
__global__ void gemm_wmma_f16(const float* __restrict__ A,
                              const float* __restrict__ B,
                              float* __restrict__ C,
                              int M, int N, int K,
                              long long strideB, long long strideC)
{
    __shared__ _Float16 Ah[TILE_M][TILE_K + APAD];
    __shared__ _Float16 Bt[TILE_N][TILE_K + APAD];   // transposed tile

    const int tid  = threadIdx.x;
    const int lane = tid & 31;
    const int wv   = tid >> 5;      // wave 0..3 -> M subtile
    const int mrow = lane & 15;
    const int hi   = lane >> 4;     // 0/1 half-wave

    const int m0 = blockIdx.y * TILE_M;
    const int n0 = blockIdx.x * TILE_N;

    B += (size_t)blockIdx.z * (size_t)strideB;
    C += (size_t)blockIdx.z * (size_t)strideC;

    v8f zero = {};
    v8f acc[4] = {zero, zero, zero, zero};

    for (int k0 = 0; k0 < K; k0 += TILE_K) {
        // ---- stage A tile 64x32: float4 loads, packed b64 LDS stores ----
        #pragma unroll
        for (int j = tid; j < (TILE_M * TILE_K) / 4; j += 128) {
            int row = j >> 3;              // 8 float4 per 32-wide row
            int c4  = (j & 7) * 4;
            v4f f = *(const v4f*)&A[(size_t)(m0 + row) * K + (k0 + c4)];
            v4h hv;
            hv[0] = (_Float16)f[0]; hv[1] = (_Float16)f[1];
            hv[2] = (_Float16)f[2]; hv[3] = (_Float16)f[3];
            *(v4h*)&Ah[row][c4] = hv;      // 8-B aligned (pitch 80 B)
        }
        // ---- stage B tile 32x64: float4 loads, transposed scalar stores ----
        #pragma unroll
        for (int j = tid; j < (TILE_K * TILE_N) / 4; j += 128) {
            int row = j >> 4;              // k (16 float4 per 64-wide row)
            int c4  = (j & 15) * 4;        // n
            v4f f = *(const v4f*)&B[(size_t)(k0 + row) * N + (n0 + c4)];
            Bt[c4 + 0][row] = (_Float16)f[0];
            Bt[c4 + 1][row] = (_Float16)f[1];
            Bt[c4 + 2][row] = (_Float16)f[2];
            Bt[c4 + 3][row] = (_Float16)f[3];
        }
        // prefetch next B tile (-> global_prefetch_b8)
        if (k0 + TILE_K < K)
            __builtin_prefetch(&B[(size_t)(k0 + TILE_K + (tid >> 6)) * N + n0 + (tid & 63)], 0, 1);
        __syncthreads();

        // ---- A fragment: halves 0..7 -> K = 8*hi + h, 8..15 -> K = 16+8*hi+h
        //      both runs contiguous & 16-B aligned -> ds_load_b128 x2 ----
        const _Float16* arow = &Ah[wv * 16 + mrow][0];
        v8h a0 = *(const v8h*)(arow + hi * 8);
        v8h a1 = *(const v8h*)(arow + 16 + hi * 8);
        v16h afrag = __builtin_shufflevector(a0, a1,
            0, 1, 2, 3, 4, 5, 6, 7, 8, 9, 10, 11, 12, 13, 14, 15);

        // ---- 4 N-subtiles: B fragment = halves K = 8*hi..+7 and 16+8*hi..+7 of Bt[n] ----
        #pragma unroll
        for (int nt = 0; nt < 4; ++nt) {
            const _Float16* brow = &Bt[nt * 16 + mrow][0];
            v8h b0 = *(const v8h*)(brow + hi * 8);
            v8h b1 = *(const v8h*)(brow + 16 + hi * 8);
            v16h bfrag = __builtin_shufflevector(b0, b1,
                0, 1, 2, 3, 4, 5, 6, 7, 8, 9, 10, 11, 12, 13, 14, 15);
            acc[nt] = __builtin_amdgcn_wmma_f32_16x16x32_f16(
                false, afrag, false, bfrag, (short)0, acc[nt], false, false);
        }
        __syncthreads();
    }

    // ---- store per C/D 16x16 f32 layout: M = 8*hi + r, N = lane&15 ----
    // (for fixed r, 16 lanes write 16 consecutive columns -> coalesced)
    #pragma unroll
    for (int nt = 0; nt < 4; ++nt) {
        int col = n0 + nt * 16 + mrow;
        #pragma unroll
        for (int r = 0; r < 8; ++r) {
            int row = m0 + wv * 16 + hi * 8 + r;
            C[(size_t)row * N + col] = acc[nt][r];
        }
    }
}

// ---------------------------------------------------------------------------
// Depthwise 3x3, SAME (zero) padding, cross-correlation (XLA semantics).
// ---------------------------------------------------------------------------
__global__ void dwconv3x3(const float* __restrict__ in,
                          const float* __restrict__ wgt,
                          float* __restrict__ out,
                          int B, int C, int H, int W)
{
    long long idx = (long long)blockIdx.x * blockDim.x + threadIdx.x;
    long long total = (long long)B * C * H * W;
    if (idx >= total) return;
    int x  = (int)(idx % W);
    int yy = (int)((idx / W) % H);
    int ch = (int)((idx / ((long long)W * H)) % C);
    long long plane = idx / ((long long)W * H); // b*C + ch
    const float* ip = in + plane * (long long)H * W;
    const float* wp = wgt + (size_t)ch * 9;
    float s = 0.f;
    #pragma unroll
    for (int dy = -1; dy <= 1; ++dy) {
        int y2 = yy + dy;
        if (y2 < 0 || y2 >= H) continue;
        #pragma unroll
        for (int dx = -1; dx <= 1; ++dx) {
            int x2 = x + dx;
            if (x2 < 0 || x2 >= W) continue;
            s += ip[(size_t)y2 * W + x2] * wp[(dy + 1) * 3 + (dx + 1)];
        }
    }
    out[idx] = s;
}

// ---------------------------------------------------------------------------
// Row-wise L2 normalize (F.normalize over last dim, eps 1e-12).
// One 256-thread block per row. Rows addressed via per-batch strides.
// ---------------------------------------------------------------------------
__global__ void l2norm_rows(const float* __restrict__ in,
                            float* __restrict__ out,
                            int rowsPerBatch,
                            long long inBatchStride, long long outBatchStride,
                            int n)
{
    int r  = blockIdx.x;
    int b  = r / rowsPerBatch;
    int ch = r % rowsPerBatch;
    const float* ip = in  + (size_t)b * inBatchStride  + (size_t)ch * n;
    float*       op = out + (size_t)b * outBatchStride + (size_t)ch * n;

    __shared__ float red[256];
    float s = 0.f;
    for (int i = threadIdx.x; i < n; i += 256) { float v = ip[i]; s += v * v; }
    red[threadIdx.x] = s;
    __syncthreads();
    for (int off = 128; off > 0; off >>= 1) {
        if ((int)threadIdx.x < off) red[threadIdx.x] += red[threadIdx.x + off];
        __syncthreads();
    }
    float inv = 1.0f / fmaxf(sqrtf(red[0]), 1e-12f);
    for (int i = threadIdx.x; i < n; i += 256) op[i] = ip[i] * inv;
}

// ---------------------------------------------------------------------------
// Attention scores: S[b,h,c,d] = temp[h] * dot(qn[b,h,c,:], kn[b,h,d,:])
// One wave (32 lanes) per output element; qn/kn compact [b,192,n].
// ---------------------------------------------------------------------------
__global__ void attn_scores(const float* __restrict__ qn,
                            const float* __restrict__ kn,
                            const float* __restrict__ temp,
                            float* __restrict__ S, int n)
{
    int gw   = blockIdx.x * (blockDim.x >> 5) + (threadIdx.x >> 5);
    int lane = threadIdx.x & 31;
    int d = gw % 48;
    int c = (gw / 48) % 48;
    int h = (gw / (48 * 48)) % 4;
    int b = gw / (48 * 48 * 4);
    const float* qp = qn + ((size_t)b * 192 + h * 48 + c) * (size_t)n;
    const float* kp = kn + ((size_t)b * 192 + h * 48 + d) * (size_t)n;
    float s = 0.f;
    for (int i = lane; i < n; i += 32) s += qp[i] * kp[i];
    #pragma unroll
    for (int off = 16; off > 0; off >>= 1) s += __shfl_xor(s, off, 32);
    if (lane == 0) S[gw] = s * temp[h];
}

// ---------------------------------------------------------------------------
// Softmax over last dim (48). One thread per row, in place.
// ---------------------------------------------------------------------------
__global__ void softmax48(float* __restrict__ S, int rows)
{
    int r = blockIdx.x * blockDim.x + threadIdx.x;
    if (r >= rows) return;
    float* p = S + (size_t)r * 48;
    float m = -1e30f;
    for (int j = 0; j < 48; ++j) m = fmaxf(m, p[j]);
    float sum = 0.f;
    for (int j = 0; j < 48; ++j) { float e = expf(p[j] - m); p[j] = e; sum += e; }
    float inv = 1.0f / sum;
    for (int j = 0; j < 48; ++j) p[j] *= inv;
}

// ---------------------------------------------------------------------------
// out[b, h*48+c, p] = sum_d attn[b,h,c,d] * v[b, h*48+d, p]
// v addressed with a per-batch stride (block offset pre-applied by caller).
// Output compact [b,192,n].
// ---------------------------------------------------------------------------
__global__ void attn_apply(const float* __restrict__ attn,
                           const float* __restrict__ v,
                           float* __restrict__ out,
                           long long vBatchStride, int n)
{
    long long idx = (long long)blockIdx.x * blockDim.x + threadIdx.x;
    long long total = 2LL * 192 * n;
    if (idx >= total) return;
    int p  = (int)(idx % n);
    int ch = (int)((idx / n) % 192);
    int b  = (int)(idx / ((long long)n * 192));
    int h  = ch / 48, c = ch % 48;
    const float* ar   = attn + (((size_t)b * 4 + h) * 48 + c) * 48;
    const float* vrow = v + (size_t)b * vBatchStride + (size_t)h * 48 * n;
    float s = 0.f;
    #pragma unroll 4
    for (int d = 0; d < 48; ++d) s += ar[d] * vrow[(size_t)d * n + p];
    out[idx] = s;
}

// ---------------------------------------------------------------------------
extern "C" void kernel_launch(void* const* d_in, const int* in_sizes, int n_in,
                              void* d_out, int out_size, void* d_ws, size_t ws_size,
                              hipStream_t stream)
{
    const float* x      = (const float*)d_in[0];
    const float* y      = (const float*)d_in[1];
    const float* qkv_w  = (const float*)d_in[2];
    const float* dw_w   = (const float*)d_in[3];
    const float* kv_w   = (const float*)d_in[4];
    const float* proj_w = (const float*)d_in[5];
    const float* temp   = (const float*)d_in[6];
    float* out = (float*)d_out;

    const int Bn = 2, Cc = 192, H = 128, W = 128;
    const int n  = H * W;                       // 16384
    const size_t t1 = (size_t)Cc * n;           // one batch's 192xN tensor

    // Workspace layout (floats):
    //  A0: 6*t1  qkv_pre, later reused as 3 compact slots (2*t1 each)
    //  A1: 6*t1  qkv (post dwconv)
    //  A2: 4*t1  kv, later reused for out2 (2*t1)
    //  Sb: scores 2*4*48*48
    float* ws = (float*)d_ws;
    float* A0 = ws;
    float* A1 = A0 + 6 * t1;
    float* A2 = A1 + 6 * t1;
    float* Sb = A2 + 4 * t1;
    float* slot0 = A0;               // qn  -> q1n
    float* slot1 = A0 + 2 * t1;      // k2n -> kn
    float* slot2 = A0 + 4 * t1;      // q1

    dim3 blk128(128), blk256(256);

    // 1) qkv_pre = qkv_w(576x192) * x  (per batch)
    gemm_wmma_f16<<<dim3(n / 64, 576 / 64, Bn), blk128, 0, stream>>>(
        qkv_w, x, A0, 576, n, Cc, (long long)Cc * n, 3LL * Cc * n);

    // 2) qkv = dwconv3x3(qkv_pre)
    {
        long long total = (long long)Bn * 3 * Cc * n;
        dwconv3x3<<<dim3((unsigned)((total + 255) / 256)), blk256, 0, stream>>>(
            A0, dw_w, A1, Bn, 3 * Cc, H, W);
    }

    // 3) kv = kv_w(384x192) * y  (per batch)
    gemm_wmma_f16<<<dim3(n / 64, 384 / 64, Bn), blk128, 0, stream>>>(
        kv_w, y, A2, 384, n, Cc, (long long)Cc * n, 2LL * Cc * n);

    // ---- MDTA stage 1: q (from qkv) vs k2/v2 (from kv) ----
    l2norm_rows<<<dim3(Bn * Cc), blk256, 0, stream>>>(
        A1, slot0, Cc, 3LL * Cc * n, (long long)Cc * n, n);              // qn
    l2norm_rows<<<dim3(Bn * Cc), blk256, 0, stream>>>(
        A2, slot1, Cc, 2LL * Cc * n, (long long)Cc * n, n);              // k2n

    attn_scores<<<dim3((Bn * 4 * 48 * 48) / 4), blk128, 0, stream>>>(
        slot0, slot1, temp, Sb, n);
    softmax48<<<dim3(2), blk256, 0, stream>>>(Sb, Bn * 4 * 48);
    {
        long long total = 2LL * Cc * n;
        attn_apply<<<dim3((unsigned)((total + 255) / 256)), blk256, 0, stream>>>(
            Sb, A2 + (size_t)Cc * n, slot2, 2LL * Cc * n, n);            // q1 = attn1 * v2
    }

    // ---- MDTA stage 2: q1 vs k/v (from qkv) ----
    l2norm_rows<<<dim3(Bn * Cc), blk256, 0, stream>>>(
        slot2, slot0, Cc, (long long)Cc * n, (long long)Cc * n, n);      // q1n
    l2norm_rows<<<dim3(Bn * Cc), blk256, 0, stream>>>(
        A1 + (size_t)Cc * n, slot1, Cc, 3LL * Cc * n, (long long)Cc * n, n); // kn
    attn_scores<<<dim3((Bn * 4 * 48 * 48) / 4), blk128, 0, stream>>>(
        slot0, slot1, temp, Sb, n);
    softmax48<<<dim3(2), blk256, 0, stream>>>(Sb, Bn * 4 * 48);
    {
        long long total = 2LL * Cc * n;
        attn_apply<<<dim3((unsigned)((total + 255) / 256)), blk256, 0, stream>>>(
            Sb, A1 + (size_t)(2 * Cc) * n, A2, 3LL * Cc * n, n);         // out2 -> A2
    }

    // 4) final = proj_w(192x192) * out2  -> d_out
    gemm_wmma_f16<<<dim3(n / 64, Cc / 64, Bn), blk128, 0, stream>>>(
        proj_w, A2, out, Cc, n, Cc, (long long)Cc * n, (long long)Cc * n);
}
